// TSwinTransformer_15874199126216
// MI455X (gfx1250) — matrix operand order repaired and checked by reading, not compiled, not run on hardware
//
#include <hip/hip_runtime.h>

#define DIM   256
#define NH    8
#define HD    32
#define NTOK  98
#define NPAD  112
#define BW    2048
#define ROWS  (BW * NTOK)   // 200704 = 128 * 1568
#define TBL   507           // (2*2-1)*(2*7-1)*(2*7-1)

static_assert(ROWS % 128 == 0, "row tiling must be exact");

typedef _Float16 v8h  __attribute__((ext_vector_type(8)));
typedef _Float16 v16h __attribute__((ext_vector_type(16)));
typedef float    v8f  __attribute__((ext_vector_type(8)));

// Concatenate two 16-byte LDS chunks into one 16-half WMMA fragment register set.
__device__ __forceinline__ v16h frag16(const _Float16* lo, const _Float16* hi) {
  v8h a = *(const v8h*)lo;
  v8h b = *(const v8h*)hi;
  return __builtin_shufflevector(a, b, 0,1,2,3,4,5,6,7,8,9,10,11,12,13,14,15);
}

// ---------------------------------------------------------------------------
// gfx1250 async global->LDS copy (ISA 15.18.3 GLOBAL_LOAD_ASYNC_TO_LDS_B128).
// Per active lane: LDS[vdst] = MEM[vaddr], 16 bytes, tracked with ASYNCcnt.
// Generic shared pointers carry the LDS byte offset in their low 32 bits.
// ---------------------------------------------------------------------------
__device__ __forceinline__ void cp_async_b128(void* lds, const void* gptr) {
  unsigned ldsoff = (unsigned)(size_t)lds;
  unsigned long long ga = (unsigned long long)(size_t)gptr;
  asm volatile("global_load_async_to_lds_b128 %0, %1, off"
               :: "v"(ldsoff), "v"(ga) : "memory");
}
#define ASYNC_WAIT(N) asm volatile("s_wait_asynccnt %0" :: "n"(N) : "memory")

// ---------------------------------------------------------------------------
// Kernel 0: fp32 -> f16 conversion (x, qkv_w, proj_w), so every GEMM stage
// is a pure copy that can ride the async-to-LDS path.
// ---------------------------------------------------------------------------
__global__ void k_cvt(const float* __restrict__ src, _Float16* __restrict__ dst, long n4) {
  long stride = (long)gridDim.x * blockDim.x;
  for (long i = (long)blockIdx.x * blockDim.x + threadIdx.x; i < n4; i += stride) {
    float4 v = *(const float4*)(src + i * 4);
    _Float16* d = dst + i * 4;
    d[0] = (_Float16)v.x; d[1] = (_Float16)v.y;
    d[2] = (_Float16)v.z; d[3] = (_Float16)v.w;
  }
}

// ---------------------------------------------------------------------------
// Kernel 1a: continuous position bias MLP -> bt[TBL][NH]
// ---------------------------------------------------------------------------
__global__ void k_cpb_mlp(const float* __restrict__ w1, const float* __restrict__ b1,
                          const float* __restrict__ w2, float* __restrict__ bt) {
  __shared__ float hidden[512];
  int t = blockIdx.x;
  int a = t / 169, rm = t % 169, b = rm / 13, c = rm % 13;
  float g0 = (float)(a - 1) / (1.0f + 1e-6f) * 8.0f;
  float g1 = (float)(b - 6) / (6.0f + 1e-6f) * 8.0f;
  float g2 = (float)(c - 6) / (6.0f + 1e-6f) * 8.0f;
  g0 = copysignf(log2f(fabsf(g0) + 1.0f), g0) * (1.0f / 3.0f);
  g1 = copysignf(log2f(fabsf(g1) + 1.0f), g1) * (1.0f / 3.0f);
  g2 = copysignf(log2f(fabsf(g2) + 1.0f), g2) * (1.0f / 3.0f);
  for (int j = threadIdx.x; j < 512; j += 64) {
    float hv = g0 * w1[j * 3 + 0] + g1 * w1[j * 3 + 1] + g2 * w1[j * 3 + 2] + b1[j];
    hidden[j] = fmaxf(hv, 0.0f);
  }
  __syncthreads();
  if (threadIdx.x < NH) {
    int h = threadIdx.x;
    float s = 0.0f;
    for (int j = 0; j < 512; ++j) s += hidden[j] * w2[h * 512 + j];
    bt[t * NH + h] = s;
  }
}

// ---------------------------------------------------------------------------
// Kernel 1b: rpb[h][i][j] = 16*sigmoid(bt[relposindex(i,j)][h])
// ---------------------------------------------------------------------------
__global__ void k_rpb(const float* __restrict__ bt, float* __restrict__ rpb) {
  int id = blockIdx.x * 256 + threadIdx.x;
  if (id >= NH * NTOK * NTOK) return;
  int h = id / (NTOK * NTOK);
  int rem = id % (NTOK * NTOK);
  int i = rem / NTOK, j = rem % NTOK;
  int di = i / 49, hi = (i / 7) % 7, wi = i % 7;
  int dj = j / 49, hj = (j / 7) % 7, wj = j % 7;
  int idx = (di - dj + 1) * 169 + (hi - hj + 6) * 13 + (wi - wj + 6);
  float v = bt[idx * NH + h];
  rpb[id] = 16.0f / (1.0f + __expf(-v));
}

// ---------------------------------------------------------------------------
// WMMA k-step: preload all 8 B fragments, then 8 back-to-back WMMAs so the
// scheduler can use partial dscnt waits instead of full drains per WMMA.
// As/Bs: [128][CSTR] f16, Bs row = output column (B column-major contiguous).
// ---------------------------------------------------------------------------
#define CSTR 64

__device__ __forceinline__ void gemm_kstep(const _Float16* __restrict__ As,
                                           const _Float16* __restrict__ Bs,
                                           int k0, v8f acc[8]) {
  int lane = threadIdx.x & 31;
  int wave = threadIdx.x >> 5;
  int mrow = wave * 16 + (lane & 15);
  int kb = (lane >> 4) * 8;    // A half-select
  int kh = (lane >> 4) * 16;   // unused: B also uses kb
  int nc = lane & 15;
  const _Float16* ap = As + mrow * CSTR + k0;
  v16h a = frag16(ap + kb, ap + kb + 16);
  v16h bf[8];
#pragma unroll
  for (int c = 0; c < 8; ++c) {
    const _Float16* bp = Bs + (c * 16 + nc) * CSTR + k0 + kb;
    bf[c] = frag16(bp, bp + 16);
  }
#pragma unroll
  for (int c = 0; c < 8; ++c)
    acc[c] = __builtin_amdgcn_wmma_f32_16x16x32_f16(false, a, false, bf[c],
                                                    (short)0, acc[c], false, false);
}

// Double-buffered async staging of one 128x64 f16 A chunk + B chunk.
__device__ __forceinline__ void stage_chunk(_Float16* As, _Float16* Bs,
                                            const _Float16* __restrict__ ab,
                                            const _Float16* __restrict__ bb,
                                            int kc, int tid) {
#pragma unroll
  for (int it = 0; it < 4; ++it) {           // 1024 b128 chunks per matrix
    int e = tid + it * 256;
    int r = e >> 3, cj = e & 7;
    cp_async_b128(As + r * CSTR + cj * 8, ab + (long)r * 256 + kc * 64 + cj * 8);
    cp_async_b128(Bs + r * CSTR + cj * 8, bb + (long)r * 256 + kc * 64 + cj * 8);
  }
}

// ---------------------------------------------------------------------------
// Kernel 2: QKV GEMM [ROWS,256] x [256,768]^T + bias, fused per-head L2 norm
// for q/k, f16 outputs laid out [B][NH][98][32].
// grid = (ROWS/128, 6), block = 256 (8 wave32)
// ---------------------------------------------------------------------------
__global__ void __launch_bounds__(256) k_qkv(const _Float16* __restrict__ xh,
                                             const _Float16* __restrict__ qwh,
                                             const float* __restrict__ q_bias,
                                             const float* __restrict__ v_bias,
                                             _Float16* __restrict__ qbuf,
                                             _Float16* __restrict__ kbuf,
                                             _Float16* __restrict__ vbuf) {
  __shared__ __align__(16) unsigned char smem[65536];   // 2x (As 16K + Bs 16K)
  int tid = threadIdx.x, bx = blockIdx.x, by = blockIdx.y;
  const _Float16* ab = xh + (long)bx * 128 * 256;
  const _Float16* bb = qwh + (long)by * 128 * 256;

  v8f acc[8] = {};
  stage_chunk((_Float16*)smem, (_Float16*)(smem + 16384), ab, bb, 0, tid);
  for (int kc = 0; kc < 4; ++kc) {
    if (kc < 3) {
      int nbuf = (kc + 1) & 1;
      stage_chunk((_Float16*)(smem + nbuf * 32768),
                  (_Float16*)(smem + 16384 + nbuf * 32768), ab, bb, kc + 1, tid);
      ASYNC_WAIT(8);                          // allow next chunk's 8 ops in flight
    } else {
      ASYNC_WAIT(0);
    }
    __syncthreads();
    int buf = kc & 1;
    const _Float16* Ac = (const _Float16*)(smem + buf * 32768);
    const _Float16* Bc = (const _Float16*)(smem + 16384 + buf * 32768);
    gemm_kstep(Ac, Bc, 0, acc);
    gemm_kstep(Ac, Bc, 32, acc);
    __syncthreads();                          // computed buffer free for prefetch
  }

  // Epilogue: bias + per-head L2 normalize (q,k), store f16 [B][NH][98][32]
  int lane = tid & 31, wave = tid >> 5;
  int nc = lane & 15, rsel = lane >> 4;
#pragma unroll
  for (int ch = 0; ch < 4; ++ch) {            // 4 head-chunks of 32 cols
    int c0 = 2 * ch, c1 = 2 * ch + 1;
    int col0 = by * 128 + ch * 32;
    float b0 = 0.0f, b1v = 0.0f;
    if (by < 2)      { b0 = q_bias[col0 + nc];       b1v = q_bias[col0 + 16 + nc]; }
    else if (by >= 4){ b0 = v_bias[col0 - 512 + nc]; b1v = v_bias[col0 - 512 + 16 + nc]; }
    _Float16* dstbuf; int h;
    if (by < 2)      { dstbuf = qbuf; h = by * 4 + ch; }
    else if (by < 4) { dstbuf = kbuf; h = (by - 2) * 4 + ch; }
    else             { dstbuf = vbuf; h = (by - 4) * 4 + ch; }
#pragma unroll
    for (int r = 0; r < 8; ++r) {
      float u0 = acc[c0][r] + b0;
      float u1 = acc[c1][r] + b1v;
      float ss = u0 * u0 + u1 * u1;
      for (int m = 1; m < 16; m <<= 1) ss += __shfl_xor(ss, m, 32);
      float inv = (by < 4) ? (1.0f / fmaxf(sqrtf(ss), 1e-12f)) : 1.0f;
      long row_g = (long)bx * 128 + wave * 16 + r + 8 * rsel;
      int b = (int)(row_g / NTOK), tok = (int)(row_g % NTOK);
      _Float16* dst = dstbuf + (((long)b * NH + h) * NTOK + tok) * HD;
      dst[nc]      = (_Float16)(u0 * inv);
      dst[16 + nc] = (_Float16)(u1 * inv);
    }
  }
}

// ---------------------------------------------------------------------------
// Kernel 3: per-(window,head) attention. grid = BW*NH, block = 256.
// S = Q K^T * scale + rpb (mask pads), softmax, P (f16, in-place over S), P@V.
// ---------------------------------------------------------------------------
#define QSTR 32    // q/k LDS stride (halfs)
#define SSTR 112   // S stride (floats); P stride = 224 halfs
#define VSTR 128   // Vt stride (halfs)

__global__ void __launch_bounds__(256) k_attn(const _Float16* __restrict__ qbuf,
                                              const _Float16* __restrict__ kbuf,
                                              const _Float16* __restrict__ vbuf,
                                              const float* __restrict__ rpb,
                                              const float* __restrict__ logit_scale,
                                              _Float16* __restrict__ attn_out) {
  __shared__ __align__(16) unsigned char smem[NPAD * QSTR * 2 * 2 + NPAD * SSTR * 4]; // 64512 B
  _Float16* Qs = (_Float16*)smem;                       // [112][32]
  _Float16* Ks = (_Float16*)(smem + NPAD * QSTR * 2);   // [112][32]
  _Float16* Vt = (_Float16*)smem;                       // [32][128], reuses Q/K after S
  float*    S  = (float*)(smem + 2 * NPAD * QSTR * 2);  // [112][112]

  int tid = threadIdx.x;
  int b = blockIdx.x >> 3, h = blockIdx.x & 7;
  float lsc = __expf(fminf(logit_scale[h], 4.6051702f));   // clamp at log(100)

  const _Float16* qsrc = qbuf + ((long)b * NH + h) * NTOK * HD;
  const _Float16* ksrc = kbuf + ((long)b * NH + h) * NTOK * HD;
  v8h z8 = {};
  for (int e = tid; e < NPAD * 4; e += 256) {  // 112 rows x 4 b128 chunks
    int r = e >> 2, cj = e & 3;
    if (r < NTOK) {
      cp_async_b128(Qs + r * QSTR + cj * 8, qsrc + r * HD + cj * 8);
      cp_async_b128(Ks + r * QSTR + cj * 8, ksrc + r * HD + cj * 8);
    } else {
      *(v8h*)(Qs + r * QSTR + cj * 8) = z8;
      *(v8h*)(Ks + r * QSTR + cj * 8) = z8;
    }
  }
  ASYNC_WAIT(0);
  __syncthreads();

  int lane = tid & 31, wave = tid >> 5;
  int kb = (lane >> 4) * 8, kh = (lane >> 4) * 16;
  int mr = lane & 15, nc = lane & 15, rsel = lane >> 4;
  const float* rp = rpb + h * NTOK * NTOK;

  for (int t = wave; t < 49; t += 8) {       // S tiles: K = 32 = hd in one WMMA
    int ti = t / 7, tj = t % 7;
    const _Float16* ap = Qs + (ti * 16 + mr) * QSTR;
    v16h a = frag16(ap + kb, ap + kb + 16);
    const _Float16* bp = Ks + (tj * 16 + nc) * QSTR + kb;
    v16h bfr = frag16(bp, bp + 16);
    v8f c = {};
    c = __builtin_amdgcn_wmma_f32_16x16x32_f16(false, a, false, bfr, (short)0, c, false, false);
#pragma unroll
    for (int r = 0; r < 8; ++r) {
      int i = ti * 16 + r + 8 * rsel;
      int j = tj * 16 + nc;
      float v = -1e30f;
      if (i < NTOK && j < NTOK) v = c[r] * lsc + rp[i * NTOK + j];
      S[i * SSTR + j] = v;
    }
  }
  __syncthreads();

  // Stage Vt[d][tok] (transposed, zero-pad tokens) into freed Q/K space.
  const _Float16* vsrc = vbuf + ((long)b * NH + h) * NTOK * HD;
#pragma unroll
  for (int it = 0; it < 16; ++it) {
    int e = tid + it * 256;                  // 32*128
    int d = e >> 7, t = e & 127;
    Vt[d * VSTR + t] = (t < NTOK) ? vsrc[t * HD + d] : (_Float16)0.0f;
  }

  // Softmax per row, pack f16 probs in place over S (thread owns whole row).
  if (tid < NPAD) {
    float* row = S + tid * SSTR;
    _Float16* prow = (_Float16*)row;         // stride 224 halfs
    if (tid < NTOK) {
      float m = -1e30f;
      for (int j = 0; j < NTOK; ++j) m = fmaxf(m, row[j]);
      float sum = 0.0f;
      for (int j = 0; j < NTOK; ++j) sum += __expf(row[j] - m);
      float inv = 1.0f / sum;
      for (int j = 0; j < NTOK; ++j) {
        float p = __expf(row[j] - m) * inv;  // reads float j, writes half j (j/2 already read)
        prow[j] = (_Float16)p;
      }
      for (int j = NTOK; j < 128; ++j) prow[j] = (_Float16)0.0f;
    } else {
      for (int j = 0; j < 128; ++j) prow[j] = (_Float16)0.0f;
    }
  }
  __syncthreads();

  // P @ V : 7 row tiles x 2 hd-col tiles, K = 128 (padded) in 4 WMMA steps.
  const _Float16* P = (const _Float16*)S;    // [112][224] halfs, cols 0..127 valid
  for (int t = wave; t < 14; t += 8) {
    int ti = t >> 1, ct = t & 1;
    v8f acc = {};
#pragma unroll
    for (int ks = 0; ks < 4; ++ks) {
      int k0 = ks * 32;
      const _Float16* ap = P + (ti * 16 + mr) * (2 * SSTR) + k0;
      v16h a = frag16(ap + kb, ap + kb + 16);
      const _Float16* bp = Vt + (ct * 16 + nc) * VSTR + k0 + kb;
      v16h bfr = frag16(bp, bp + 16);
      acc = __builtin_amdgcn_wmma_f32_16x16x32_f16(false, a, false, bfr, (short)0, acc, false, false);
    }
#pragma unroll
    for (int r = 0; r < 8; ++r) {
      int tok = ti * 16 + r + 8 * rsel;
      if (tok < NTOK) {
        int d = ct * 16 + nc;
        attn_out[((long)b * NTOK + tok) * DIM + h * HD + d] = (_Float16)acc[r];
      }
    }
  }
}

// ---------------------------------------------------------------------------
// Kernel 4: projection GEMM [ROWS,256] x [256,256]^T + bias -> fp32 out.
// grid = (ROWS/128, 2), block = 256
// ---------------------------------------------------------------------------
__global__ void __launch_bounds__(256) k_proj(const _Float16* __restrict__ ain,
                                              const _Float16* __restrict__ pwh,
                                              const float* __restrict__ proj_b,
                                              float* __restrict__ out) {
  __shared__ __align__(16) unsigned char smem[65536];
  int tid = threadIdx.x, bx = blockIdx.x, by = blockIdx.y;
  const _Float16* ab = ain + (long)bx * 128 * 256;
  const _Float16* bb = pwh + (long)by * 128 * 256;

  v8f acc[8] = {};
  stage_chunk((_Float16*)smem, (_Float16*)(smem + 16384), ab, bb, 0, tid);
  for (int kc = 0; kc < 4; ++kc) {
    if (kc < 3) {
      int nbuf = (kc + 1) & 1;
      stage_chunk((_Float16*)(smem + nbuf * 32768),
                  (_Float16*)(smem + 16384 + nbuf * 32768), ab, bb, kc + 1, tid);
      ASYNC_WAIT(8);
    } else {
      ASYNC_WAIT(0);
    }
    __syncthreads();
    int buf = kc & 1;
    const _Float16* Ac = (const _Float16*)(smem + buf * 32768);
    const _Float16* Bc = (const _Float16*)(smem + 16384 + buf * 32768);
    gemm_kstep(Ac, Bc, 0, acc);
    gemm_kstep(Ac, Bc, 32, acc);
    __syncthreads();
  }

  int lane = tid & 31, wave = tid >> 5;
  int nc = lane & 15, rsel = lane >> 4;
#pragma unroll
  for (int c = 0; c < 8; ++c) {
    int col_g = by * 128 + c * 16 + nc;
    float bias = proj_b[col_g];
#pragma unroll
    for (int r = 0; r < 8; ++r) {
      long row_g = (long)bx * 128 + wave * 16 + r + 8 * rsel;
      out[row_g * 256 + col_g] = acc[c][r] + bias;
    }
  }
}

// ---------------------------------------------------------------------------
// Host launcher. Workspace: f16 copies of x/weights + bt + rpb + q/k/v + attn.
// ---------------------------------------------------------------------------
extern "C" void kernel_launch(void* const* d_in, const int* in_sizes, int n_in,
                              void* d_out, int out_size, void* d_ws, size_t ws_size,
                              hipStream_t stream) {
  (void)in_sizes; (void)n_in; (void)out_size; (void)ws_size;
  const float* x     = (const float*)d_in[0];
  const float* qkvw  = (const float*)d_in[1];
  const float* qbias = (const float*)d_in[2];
  const float* vbias = (const float*)d_in[3];
  const float* lsc   = (const float*)d_in[4];
  const float* w1    = (const float*)d_in[5];
  const float* b1    = (const float*)d_in[6];
  const float* w2    = (const float*)d_in[7];
  const float* pw    = (const float*)d_in[8];
  const float* pb    = (const float*)d_in[9];
  float* out = (float*)d_out;

  char* ws = (char*)d_ws;
  size_t off = 0;
  auto take = [&](size_t bytes) {
    char* p = ws + off;
    off = (off + bytes + 255) & ~(size_t)255;
    return p;
  };
  float*    bt   = (float*)take((size_t)TBL * NH * sizeof(float));
  float*    rpb  = (float*)take((size_t)NH * NTOK * NTOK * sizeof(float));
  _Float16* xh   = (_Float16*)take((size_t)ROWS * DIM * sizeof(_Float16));
  _Float16* qwh  = (_Float16*)take((size_t)3 * DIM * DIM * sizeof(_Float16));
  _Float16* pwh  = (_Float16*)take((size_t)DIM * DIM * sizeof(_Float16));
  size_t    qkvb = (size_t)BW * NH * NTOK * HD * sizeof(_Float16);
  _Float16* qbuf = (_Float16*)take(qkvb);
  _Float16* kbuf = (_Float16*)take(qkvb);
  _Float16* vbuf = (_Float16*)take(qkvb);
  _Float16* abuf = (_Float16*)take((size_t)ROWS * DIM * sizeof(_Float16));

  hipLaunchKernelGGL(k_cvt, dim3(4096), dim3(256), 0, stream, x, xh, (long)ROWS * DIM / 4);
  hipLaunchKernelGGL(k_cvt, dim3(192), dim3(256), 0, stream, qkvw, qwh, (long)3 * DIM * DIM / 4);
  hipLaunchKernelGGL(k_cvt, dim3(64), dim3(256), 0, stream, pw, pwh, (long)DIM * DIM / 4);
  hipLaunchKernelGGL(k_cpb_mlp, dim3(TBL), dim3(64), 0, stream, w1, b1, w2, bt);
  hipLaunchKernelGGL(k_rpb, dim3((NH * NTOK * NTOK + 255) / 256), dim3(256), 0, stream, bt, rpb);
  hipLaunchKernelGGL(k_qkv, dim3(ROWS / 128, 6), dim3(256), 0, stream,
                     xh, qwh, qbias, vbias, qbuf, kbuf, vbuf);
  hipLaunchKernelGGL(k_attn, dim3(BW * NH), dim3(256), 0, stream,
                     qbuf, kbuf, vbuf, rpb, lsc, abuf);
  hipLaunchKernelGGL(k_proj, dim3(ROWS / 128, 2), dim3(256), 0, stream, abuf, pwh, pb, out);
}
